// SE3WarpingFieldEnsem_87239375716671
// MI455X (gfx1250) — matrix operand order repaired and checked by reading, not compiled, not run on hardware
//
#include <hip/hip_runtime.h>

typedef __attribute__((ext_vector_type(16))) _Float16 v16h;
typedef __attribute__((ext_vector_type(8)))  _Float16 v8h;
typedef __attribute__((ext_vector_type(4)))  _Float16 v4h;
typedef __attribute__((ext_vector_type(8)))  float    v8f;

constexpr int TPB   = 256;
constexpr int NWAVE = 8;
constexpr int S64   = 72;    // row stride (halves) for 64-wide tiles  (16B multiple)
constexpr int S128  = 136;   // row stride (halves) for 128-wide tiles (16B multiple)

// ---------------- LDS layout ----------------
// Weight region (halves):
constexpr int W0_OFF   = 0;                       // [128][S64]   K=64  (53 valid)
constexpr int W1_OFF   = W0_OFF  + 128 * S64;     // [128][S128]
constexpr int W2_OFF   = W1_OFF  + 128 * S128;
constexpr int W3_OFF   = W2_OFF  + 128 * S128;
constexpr int W4A_OFF  = W3_OFF  + 128 * S128;    // [128][S64]   W4 cols 0..52
constexpr int W4B_OFF  = W4A_OFF + 128 * S64;     // [128][S128]  W4 cols 53..180
constexpr int W5_OFF   = W4B_OFF + 128 * S128;
constexpr int WH_OFF   = W5_OFF  + 128 * S128;    // [16][S128]   rows 0-2 Wv, 3-5 Wr
constexpr int WT_HALVES = WH_OFF + 16 * S128;

constexpr int BIAS_OFF_B  = WT_HALVES * 2;               // 6*128 layer biases + 16 head biases (f32)
constexpr int NBIAS       = 6 * 128 + 16;
constexpr int BUFIN_OFF_B = BIAS_OFF_B + NBIAS * 4;      // per-wave [32][S64] input tiles
constexpr int BUFIN_H     = 32 * S64;
constexpr int BUFX_OFF_B  = BUFIN_OFF_B + NWAVE * BUFIN_H * 2;  // per-wave [32][S128] activations
constexpr int BUFX_H      = 32 * S128;
constexpr int SMEM_BYTES  = BUFX_OFF_B + NWAVE * BUFX_H * 2;    // ~325 KB of the 320 KiB WGP LDS

static_assert(BUFIN_OFF_B % 16 == 0 && BUFX_OFF_B % 16 == 0, "LDS alignment");

// ---------------- WMMA fragment helpers ----------------
// A (16x32 f16): lane L holds row M=L&15; VGPR half-pairs carry K = koff..koff+7
// (koff = (L>=16)?8:0) and K = koff+16..koff+23.
__device__ __forceinline__ v16h load_a(const _Float16* p) {
  v8h lo = *(const v8h*)p;
  v8h hi = *(const v8h*)(p + 16);
  return __builtin_shufflevector(lo, hi, 0,1,2,3,4,5,6,7,8,9,10,11,12,13,14,15);
}
// B (32x16 f16): lane L holds col N=L&15; K = koff..koff+7 and koff+16..koff+23
// (koff = (L>=16)?8:0), same as A.
__device__ __forceinline__ v16h load_b(const _Float16* p) {
  v8h lo = *(const v8h*)p;
  v8h hi = *(const v8h*)(p + 16);
  return __builtin_shufflevector(lo, hi, 0,1,2,3,4,5,6,7,8,9,10,11,12,13,14,15);
}
__device__ __forceinline__ v8f wmma16(v16h a, v16h b, v8f c) {
  return __builtin_amdgcn_wmma_f32_16x16x32_f16(false, a, false, b, (short)0, c, false, false);
}
// ReLU in one VALU op: med3(x, 0, +inf) == clamp-below-at-0
__device__ __forceinline__ _Float16 relu_h(float x) {
  return (_Float16)__builtin_amdgcn_fmed3f(x, 0.f, __builtin_inff());
}

// ---------------- weight staging: f32 global -> f16 LDS, 4-wide, zero pad ----------------
template<int LS>
__device__ __forceinline__ void stageW(const float* g, int gstride, int gcols, int rows,
                                       _Float16* l, int tid) {
  constexpr int C4 = LS / 4;            // LS is a multiple of 4
  const int total = rows * C4;
  for (int i = tid; i < total; i += TPB) {
    int r = i / C4;
    int c = (i - r * C4) * 4;
    const float* gr = g + r * gstride;
    v4h h;
#pragma unroll
    for (int j = 0; j < 4; ++j) {
      float v = (c + j < gcols) ? gr[c + j] : 0.f;
      h[j] = (_Float16)v;
    }
    *(v4h*)(l + r * LS + c) = h;        // one ds_store_b64
  }
}

// ---------------- one dense layer: out[32x128] = relu(in[32xK] @ W^T + b) ----------------
template<int NKT, int INS>
__device__ __forceinline__ void run_layer(const _Float16* in, const _Float16* w, int ws,
                                          const float* bias, _Float16* out, int lane) {
  const int mlo   = lane & 15;
  const int koffA = (lane >> 4) << 3;   // 0 or 8
  const int koffB = (lane >> 4) << 3;   // 0 or 8
  const int mb    = (lane >> 4) << 3;   // 0 or 8
  v16h a0[NKT], a1[NKT];
#pragma unroll
  for (int kt = 0; kt < NKT; ++kt) {
    a0[kt] = load_a(in + mlo        * INS + kt * 32 + koffA);
    a1[kt] = load_a(in + (mlo + 16) * INS + kt * 32 + koffA);
  }
#pragma unroll 1
  for (int nt = 0; nt < 8; ++nt) {
    const int n = nt * 16 + mlo;
    float bvl = bias[n];
    v8f d0 = {bvl, bvl, bvl, bvl, bvl, bvl, bvl, bvl};
    v8f d1 = d0;
    const _Float16* wrow = w + n * ws;
#pragma unroll
    for (int kt = 0; kt < NKT; ++kt) {
      v16h b = load_b(wrow + kt * 32 + koffB);
      d0 = wmma16(a0[kt], b, d0);
      d1 = wmma16(a1[kt], b, d1);
    }
    _Float16* o0 = out + mb * S128 + n;          // row base; r*S128 folds into DS offset
    _Float16* o1 = o0 + 16 * S128;
#pragma unroll
    for (int r = 0; r < 8; ++r) {
      o0[r * S128] = relu_h(d0[r]);
      o1[r * S128] = relu_h(d1[r]);
    }
  }
}

// layer 4 (skip): out = relu(inp @ W4a^T + x @ W4b^T + b)
__device__ __forceinline__ void run_layer4(const _Float16* bin, const _Float16* bx,
                                           const _Float16* w4a, const _Float16* w4b,
                                           const float* bias, _Float16* out, int lane) {
  const int mlo = lane & 15, koffA = (lane >> 4) << 3, koffB = (lane >> 4) << 3, mb = (lane >> 4) << 3;
  v16h p0[2], p1[2], x0[4], x1[4];
#pragma unroll
  for (int kt = 0; kt < 2; ++kt) {
    p0[kt] = load_a(bin + mlo        * S64 + kt * 32 + koffA);
    p1[kt] = load_a(bin + (mlo + 16) * S64 + kt * 32 + koffA);
  }
#pragma unroll
  for (int kt = 0; kt < 4; ++kt) {
    x0[kt] = load_a(bx + mlo        * S128 + kt * 32 + koffA);
    x1[kt] = load_a(bx + (mlo + 16) * S128 + kt * 32 + koffA);
  }
#pragma unroll 1
  for (int nt = 0; nt < 8; ++nt) {
    const int n = nt * 16 + mlo;
    float bvl = bias[n];
    v8f d0 = {bvl, bvl, bvl, bvl, bvl, bvl, bvl, bvl};
    v8f d1 = d0;
    const _Float16* wra = w4a + n * S64;
    const _Float16* wrb = w4b + n * S128;
#pragma unroll
    for (int kt = 0; kt < 2; ++kt) {
      v16h b = load_b(wra + kt * 32 + koffB);
      d0 = wmma16(p0[kt], b, d0);
      d1 = wmma16(p1[kt], b, d1);
    }
#pragma unroll
    for (int kt = 0; kt < 4; ++kt) {
      v16h b = load_b(wrb + kt * 32 + koffB);
      d0 = wmma16(x0[kt], b, d0);
      d1 = wmma16(x1[kt], b, d1);
    }
    _Float16* o0 = out + mb * S128 + n;
    _Float16* o1 = o0 + 16 * S128;
#pragma unroll
    for (int r = 0; r < 8; ++r) {
      o0[r * S128] = relu_h(d0[r]);
      o1[r * S128] = relu_h(d1[r]);
    }
  }
}

// head: screw[32x6] = feat @ Whead^T + hb   (rows 0-2 -> v, 3-5 -> r)
__device__ __forceinline__ void run_head(const _Float16* bx, const _Float16* wh,
                                         const float* hb, float* sbuf, int lane) {
  const int mlo = lane & 15, koffA = (lane >> 4) << 3, koffB = (lane >> 4) << 3, mb = (lane >> 4) << 3;
  v16h x0[4], x1[4];
#pragma unroll
  for (int kt = 0; kt < 4; ++kt) {
    x0[kt] = load_a(bx + mlo        * S128 + kt * 32 + koffA);
    x1[kt] = load_a(bx + (mlo + 16) * S128 + kt * 32 + koffA);
  }
  float bvl = hb[mlo];
  v8f d0 = {bvl, bvl, bvl, bvl, bvl, bvl, bvl, bvl};
  v8f d1 = d0;
#pragma unroll
  for (int kt = 0; kt < 4; ++kt) {
    v16h b = load_b(wh + mlo * S128 + kt * 32 + koffB);
    d0 = wmma16(x0[kt], b, d0);
    d1 = wmma16(x1[kt], b, d1);
  }
  if (mlo < 6) {
    float* s0 = sbuf + mb * 9 + mlo;
    float* s1 = s0 + 16 * 9;
#pragma unroll
    for (int r = 0; r < 8; ++r) {
      s0[r * 9] = d0[r];
      s1[r * 9] = d1[r];
    }
  }
}

__global__ void __launch_bounds__(TPB, 1)
se3_warp_kernel(const float* __restrict__ pos, const float* __restrict__ dirs,
                const float* __restrict__ wcode,
                const float* __restrict__ W0, const float* __restrict__ b0,
                const float* __restrict__ W1, const float* __restrict__ b1,
                const float* __restrict__ W2, const float* __restrict__ b2,
                const float* __restrict__ W3, const float* __restrict__ b3,
                const float* __restrict__ W4, const float* __restrict__ b4,
                const float* __restrict__ W5, const float* __restrict__ b5,
                const float* __restrict__ Wr, const float* __restrict__ br,
                const float* __restrict__ Wv, const float* __restrict__ bv,
                float* __restrict__ out, int Ntot) {
  extern __shared__ char smem[];
  _Float16* wt  = (_Float16*)smem;
  float*    sb  = (float*)(smem + BIAS_OFF_B);
  float*    hb  = sb + 6 * 128;
  _Float16* bufin_all = (_Float16*)(smem + BUFIN_OFF_B);
  _Float16* bufx_all  = (_Float16*)(smem + BUFX_OFF_B);

  const int tid = threadIdx.x;

  // ---- stage all weights to LDS f16 (zero-padded, 4-wide) ----
  stageW<S64 >(W0,       53,  53, 128, wt + W0_OFF,  tid);
  stageW<S128>(W1,      128, 128, 128, wt + W1_OFF,  tid);
  stageW<S128>(W2,      128, 128, 128, wt + W2_OFF,  tid);
  stageW<S128>(W3,      128, 128, 128, wt + W3_OFF,  tid);
  stageW<S64 >(W4,      181,  53, 128, wt + W4A_OFF, tid);
  stageW<S128>(W4 + 53, 181, 128, 128, wt + W4B_OFF, tid);
  stageW<S128>(W5,      128, 128, 128, wt + W5_OFF,  tid);
  for (int i = tid; i < 16 * (S128 / 4); i += TPB) {
    int r = i / (S128 / 4);
    int c = (i - r * (S128 / 4)) * 4;
    v4h h;
#pragma unroll
    for (int j = 0; j < 4; ++j) {
      int cc = c + j;
      float v = 0.f;
      if (cc < 128) v = (r < 3) ? Wv[r * 128 + cc] : ((r < 6) ? Wr[(r - 3) * 128 + cc] : 0.f);
      h[j] = (_Float16)v;
    }
    *(v4h*)(wt + WH_OFF + r * S128 + c) = h;
  }
  for (int i = tid; i < 128; i += TPB) {
    sb[i] = b0[i]; sb[128 + i] = b1[i]; sb[256 + i] = b2[i];
    sb[384 + i] = b3[i]; sb[512 + i] = b4[i]; sb[640 + i] = b5[i];
  }
  if (tid < 16) hb[tid] = (tid < 3) ? bv[tid] : ((tid < 6) ? br[tid - 3] : 0.f);

  // ---- per-wave input staging: NeRF encoding row (53 cols, zero-pad to 64) ----
  const int wave = tid >> 5, lane = tid & 31;
  _Float16* bin = bufin_all + wave * BUFIN_H;
  _Float16* bx  = bufx_all  + wave * BUFX_H;
  const int raw = blockIdx.x * (NWAVE * 32) + tid;
  const int idx = (raw < Ntot) ? raw : (Ntot - 1);
  {
    const float* pp = pos + (size_t)idx * 3;
    float pv[3] = {pp[0], pp[1], pp[2]};
    _Float16* row = bin + lane * S64;
#pragma unroll
    for (int d = 0; d < 3; ++d) {
      float base = 6.283185307179586f * pv[d];
#pragma unroll
      for (int f = 0; f < 7; ++f) {
        float ang = base * (float)(1 << f);
        row[d * 7 + f]      = (_Float16)__sinf(ang);
        row[21 + d * 7 + f] = (_Float16)__cosf(ang);
      }
    }
    row[42] = (_Float16)pv[0]; row[43] = (_Float16)pv[1]; row[44] = (_Float16)pv[2];
    const float* wc = wcode + (size_t)idx * 8;
#pragma unroll
    for (int j = 0; j < 8; ++j) row[45 + j] = (_Float16)wc[j];
#pragma unroll
    for (int c = 53; c < 64; ++c) row[c] = (_Float16)0.f;
  }
  __syncthreads();

  // ---- MLP stem (all GEMMs via v_wmma_f32_16x16x32_f16) ----
  run_layer<2, S64 >(bin, wt + W0_OFF, S64,  sb,       bx, lane);
  run_layer<4, S128>(bx,  wt + W1_OFF, S128, sb + 128, bx, lane);
  run_layer<4, S128>(bx,  wt + W2_OFF, S128, sb + 256, bx, lane);
  run_layer<4, S128>(bx,  wt + W3_OFF, S128, sb + 384, bx, lane);
  run_layer4(bin, bx, wt + W4A_OFF, wt + W4B_OFF, sb + 512, bx, lane);
  run_layer<4, S128>(bx,  wt + W5_OFF, S128, sb + 640, bx, lane);

  // ---- heads -> screw staging (reuse input tile LDS) ----
  float* sbuf = (float*)bin;                   // 32 rows x 9 f32, bank-conflict free
  run_head(bx, wt + WH_OFF, hb, sbuf, lane);
  asm volatile("s_wait_dscnt 0" ::: "memory"); // fence head stores before screw loads

  // ---- SE(3) exp map + apply, one point per lane ----
  float vx = sbuf[lane * 9 + 0], vy = sbuf[lane * 9 + 1], vz = sbuf[lane * 9 + 2];
  float wx = sbuf[lane * 9 + 3], wy = sbuf[lane * 9 + 4], wz = sbuf[lane * 9 + 5];
  float ang2 = fmaxf(wx * wx + wy * wy + wz * wz, 1e-4f);
  float ang  = sqrtf(ang2);
  float sa = __sinf(ang), ca = __cosf(ang);
  float f1 = sa / ang;
  float f2 = (1.f - ca) / ang2;
  float f3 = (ang - sa) / (ang * ang2);
  float Km[3][3] = {{0.f, wz, -wy}, {-wz, 0.f, wx}, {wy, -wx, 0.f}};
  float R[3][3], V[3][3];
#pragma unroll
  for (int i = 0; i < 3; ++i) {
#pragma unroll
    for (int j = 0; j < 3; ++j) {
      float kk = Km[i][0] * Km[0][j] + Km[i][1] * Km[1][j] + Km[i][2] * Km[2][j];
      float id = (i == j) ? 1.f : 0.f;
      R[i][j] = id + f1 * Km[i][j] + f2 * kk;
      V[i][j] = id + f2 * Km[i][j] + f3 * kk;
    }
  }
  float T[3];
#pragma unroll
  for (int i = 0; i < 3; ++i) T[i] = V[i][0] * vx + V[i][1] * vy + V[i][2] * vz;

  const float* pp = pos  + (size_t)idx * 3;
  const float* dd = dirs + (size_t)idx * 3;
  float p[3]  = {pp[0], pp[1], pp[2]};
  float dv[3] = {dd[0], dd[1], dd[2]};
  float wp[3], wd[3];
#pragma unroll
  for (int i = 0; i < 3; ++i) {
    wp[i] = R[0][i] * p[0]  + R[1][i] * p[1]  + R[2][i] * p[2] + T[i];  // R^T p + T
    wd[i] = R[0][i] * dv[0] + R[1][i] * dv[1] + R[2][i] * dv[2];        // R^T d
    wp[i] = (wp[i] != wp[i]) ? p[i]  : wp[i];
    wd[i] = (wd[i] != wd[i]) ? dv[i] : wd[i];
  }
  if (raw < Ntot) {
#pragma unroll
    for (int i = 0; i < 3; ++i) {
      out[(size_t)idx * 3 + i]                    = wp[i];
      out[(size_t)Ntot * 3 + (size_t)idx * 3 + i] = wd[i];
    }
  }
}

extern "C" void kernel_launch(void* const* d_in, const int* in_sizes, int n_in,
                              void* d_out, int out_size, void* d_ws, size_t ws_size,
                              hipStream_t stream) {
  (void)n_in; (void)out_size; (void)d_ws; (void)ws_size;
  const float* pos = (const float*)d_in[0];
  const float* dir = (const float*)d_in[1];
  const float* wc  = (const float*)d_in[2];
  const float* W0 = (const float*)d_in[3];  const float* b0 = (const float*)d_in[4];
  const float* W1 = (const float*)d_in[5];  const float* b1 = (const float*)d_in[6];
  const float* W2 = (const float*)d_in[7];  const float* b2 = (const float*)d_in[8];
  const float* W3 = (const float*)d_in[9];  const float* b3 = (const float*)d_in[10];
  const float* W4 = (const float*)d_in[11]; const float* b4 = (const float*)d_in[12];
  const float* W5 = (const float*)d_in[13]; const float* b5 = (const float*)d_in[14];
  const float* Wr = (const float*)d_in[15]; const float* br = (const float*)d_in[16];
  const float* Wv = (const float*)d_in[17]; const float* bv = (const float*)d_in[18];

  const int N = in_sizes[0] / 3;
  const int grid = (N + TPB - 1) / TPB;

  hipFuncSetAttribute(reinterpret_cast<const void*>(se3_warp_kernel),
                      hipFuncAttributeMaxDynamicSharedMemorySize, SMEM_BYTES);
  se3_warp_kernel<<<grid, TPB, SMEM_BYTES, stream>>>(
      pos, dir, wc, W0, b0, W1, b1, W2, b2, W3, b3, W4, b4, W5, b5,
      Wr, br, Wv, bv, (float*)d_out, N);
}
